// RelativeGlobalAttention_5531917877525
// MI455X (gfx1250) — matrix operand order repaired and checked by reading, not compiled, not run on hardware
//
#include <hip/hip_runtime.h>
#include <hip/hip_bf16.h>
#include <stdint.h>

#define BATCH 8
#define SEQ   2048
#define DIM   256

typedef __attribute__((ext_vector_type(16))) __bf16 v16bf;
typedef __attribute__((ext_vector_type(8)))  float  v8f;
typedef __attribute__((ext_vector_type(4)))  unsigned int v4ui;
typedef __attribute__((ext_vector_type(8)))  int v8si;
typedef __attribute__((ext_vector_type(4)))  int v4si;

union BF16Frag {
  v16bf v;
  uint4 q[2];
  unsigned short s[16];
};

__device__ __forceinline__ unsigned short f2bf(float f) {
  unsigned u = __float_as_uint(f);
  u += 0x7FFFu + ((u >> 16) & 1u);   // round-to-nearest-even
  return (unsigned short)(u >> 16);
}
__device__ __forceinline__ float bf2f(unsigned short h) {
  return __uint_as_float(((unsigned)h) << 16);
}

// ---- Tensor Data Mover: 2D tile (bf16 elements) global -> LDS ----
// D# per CDNA5 ISA. group0: count=1, lds_addr, 57b
// global_addr, type=2. group1: data_size=1 (2B), tensor_dim0/1, tile_dim0/1,
// tensor_dim0_stride. groups 2/3 zero (<=2D tensor). 6-arg builtin form.
__device__ __forceinline__ void tdm_load_2d(unsigned lds_addr,
                                            unsigned long long gaddr,
                                            unsigned tensor_d0, unsigned tensor_d1,
                                            unsigned tile_d0, unsigned tile_d1,
                                            unsigned long long stride0) {
  v4ui g0;
  g0[0] = 1u;                                                    // count=1
  g0[1] = lds_addr;
  g0[2] = (unsigned)gaddr;
  g0[3] = (unsigned)((gaddr >> 32) & 0x01FFFFFFu) | 0x80000000u; // type=2
  v8si g1;
  g1[0] = 0x00010000;                                            // data_size=2B
  g1[1] = (int)((tensor_d0 & 0xFFFFu) << 16);
  g1[2] = (int)((tensor_d0 >> 16) | ((tensor_d1 & 0xFFFFu) << 16));
  g1[3] = (int)((tensor_d1 >> 16) | (tile_d0 << 16));
  g1[4] = (int)(tile_d1 & 0xFFFFu);
  g1[5] = (int)(stride0 & 0xFFFFFFFFull);
  g1[6] = (int)((stride0 >> 32) & 0xFFFFull);
  g1[7] = 0;
  v4si z4 = {0, 0, 0, 0};
  v8si z8 = {0, 0, 0, 0, 0, 0, 0, 0};
  __builtin_amdgcn_tensor_load_to_lds(g0, g1, z4, z4, z8, 0);
}

__device__ __forceinline__ unsigned lds_off(const void* p) {
  return (unsigned)(unsigned long long)p;   // low 32 bits = LDS byte offset
}

// ---------------- elementwise helpers ----------------

__global__ void cvt_bf16_kernel(const float* __restrict__ in,
                                unsigned short* __restrict__ out, int n) {
  int i = blockIdx.x * blockDim.x + threadIdx.x;
  if (i < n) out[i] = f2bf(in[i]);
}

__global__ void cvt_wt_kernel(const float* __restrict__ w,
                              unsigned short* __restrict__ wt) {
  int i = blockIdx.x * blockDim.x + threadIdx.x;
  if (i >= DIM * DIM) return;
  int r = i >> 8, c = i & 255;
  wt[i] = f2bf(w[c * DIM + r]);
}

__global__ void make_e_kernel(const unsigned short* __restrict__ kb,
                              const unsigned short* __restrict__ qb,
                              unsigned short* __restrict__ eb, int n) {
  int i = blockIdx.x * blockDim.x + threadIdx.x;
  if (i < n) eb[i] = f2bf(bf2f(kb[i]) - bf2f(qb[i]));
}

// ---------------- generic NT GEMM: C[m,n] = sum_k A[m,k] * B[n,k] ----------------
// Double-buffered fragment pipeline: loads for chunk k+1 in flight during
// WMMAs of chunk k (Kd must be a multiple of 64).
__global__ __launch_bounds__(256)
void nt_gemm_bf16_kernel(const unsigned short* __restrict__ A,
                         const unsigned short* __restrict__ B,
                         unsigned short* __restrict__ C,
                         int M, int N, int Kd,
                         long long strideA, long long strideB, long long strideC,
                         int mode) {
  int lane = threadIdx.x & 31;
  int nl = lane & 15, hi = lane >> 4;
  int wave = blockIdx.x * (blockDim.x >> 5) + (threadIdx.x >> 5);
  int nTiles = N >> 6;
  int mt = wave / nTiles, nt = wave % nTiles;
  if (mt >= (M >> 4)) return;
  const unsigned short* Ab = A + (size_t)blockIdx.z * strideA;
  const unsigned short* Bb = B + (size_t)blockIdx.z * strideB;

  v8f acc[4];
#pragma unroll
  for (int t = 0; t < 4; ++t)
#pragma unroll
    for (int i = 0; i < 8; ++i) acc[t][i] = 0.f;

  const unsigned short* arow = Ab + (size_t)(mt * 16 + nl) * Kd;

  auto loadA = [&](BF16Frag& a, int kc) {
    a.q[0] = *(const uint4*)(arow + kc + hi * 8);
    a.q[1] = *(const uint4*)(arow + kc + 16 + hi * 8);
  };
  auto loadB = [&](BF16Frag* bfr, int kc) {
#pragma unroll
    for (int t = 0; t < 4; ++t) {
      const unsigned short* br =
          Bb + (size_t)(nt * 64 + t * 16 + nl) * Kd + kc + hi * 8;
      bfr[t].q[0] = *(const uint4*)(br);
      bfr[t].q[1] = *(const uint4*)(br + 16);
    }
  };
  auto domma = [&](BF16Frag& a, BF16Frag* bfr) {
#pragma unroll
    for (int t = 0; t < 4; ++t)
      acc[t] = __builtin_amdgcn_wmma_f32_16x16x32_bf16(
          false, a.v, false, bfr[t].v, (short)0, acc[t], false, false);
  };

  BF16Frag aC, aN, bC[4], bN[4];
  loadA(aC, 0);
  loadB(bC, 0);
  for (int kc = 0; kc < Kd; kc += 64) {
    loadA(aN, kc + 32);          // prefetch odd chunk
    loadB(bN, kc + 32);
    domma(aC, bC);               // consume even chunk
    if (kc + 64 < Kd) {
      loadA(aC, kc + 64);        // prefetch next even chunk
      loadB(bC, kc + 64);
    }
    domma(aN, bN);               // consume odd chunk
  }

  if (mode == 0) {
    unsigned short* Cb = C + (size_t)blockIdx.z * strideC;
#pragma unroll
    for (int t = 0; t < 4; ++t)
#pragma unroll
      for (int i = 0; i < 8; ++i)
        Cb[(size_t)(mt * 16 + hi * 8 + i) * N + nt * 64 + t * 16 + nl] =
            f2bf(acc[t][i]);
  } else {
    int mg = mt * 16 + hi * 8;
    int bidx = mg / SEQ, mloc = mg % SEQ;
#pragma unroll
    for (int t = 0; t < 4; ++t) {
      int ng = nt * 64 + t * 16 + nl;
      union { uint4 q; unsigned short s[8]; } pk;
#pragma unroll
      for (int i = 0; i < 8; ++i) pk.s[i] = f2bf(acc[t][i]);
      *(uint4*)(C + ((size_t)bidx * DIM + ng) * SEQ + mloc) = pk.q;
    }
  }
}

// ---------------- fused flash attention with skewed Srel gather ----------------
// 8 waves / block, one 16-row q-tile per wave. K and V tiles for the 32-key
// iteration are DMAd once per block into double-buffered LDS by the TDM
// (issued by wave 0), overlapping DMA of tile i+1 with compute on tile i.
__global__ __launch_bounds__(256)
void rga_attn_kernel(const unsigned short* __restrict__ Qb,   // [B,SEQ,DIM] bf16
                     const unsigned short* __restrict__ Kb,   // [B,SEQ,DIM] bf16
                     const unsigned short* __restrict__ Vt,   // [B,DIM,SEQ] bf16
                     const unsigned short* __restrict__ QE,   // [B,SEQ,SEQ] bf16
                     float* __restrict__ out) {               // [B,SEQ,DIM] f32
  __shared__ __align__(16) unsigned short ldsK[2][32 * DIM];  // 2 x 16KB
  __shared__ __align__(16) unsigned short ldsV[2][DIM * 32];  // 2 x 16KB
  __shared__ __align__(16) unsigned short pShm[8][16 * 32];   // P stage, 8KB
  int wave = threadIdx.x >> 5, lane = threadIdx.x & 31;
  int nl = lane & 15, hi = lane >> 4;
  int b = blockIdx.y;
  int q0 = blockIdx.x * 128 + wave * 16;
  const unsigned short* Q   = Qb + (size_t)b * SEQ * DIM;
  const unsigned short* Kp  = Kb + (size_t)b * SEQ * DIM;
  const unsigned short* Vp  = Vt + (size_t)b * DIM * SEQ;
  const unsigned short* QEp = QE + (size_t)b * SEQ * SEQ;

  // Preload Q A-fragments (16x256 bf16 = 64 VGPRs)
  BF16Frag aq[8];
  {
    const unsigned short* qrow = Q + (size_t)(q0 + nl) * DIM;
#pragma unroll
    for (int c = 0; c < 8; ++c) {
      aq[c].q[0] = *(const uint4*)(qrow + c * 32 + hi * 8);
      aq[c].q[1] = *(const uint4*)(qrow + c * 32 + 16 + hi * 8);
    }
  }

  v8f o[16];
#pragma unroll
  for (int t = 0; t < 16; ++t)
#pragma unroll
    for (int i = 0; i < 8; ++i) o[t][i] = 0.f;
  float mrow[8], srow[8];
#pragma unroll
  for (int i = 0; i < 8; ++i) { mrow[i] = -1e30f; srow[i] = 0.f; }

  unsigned short* pw = &pShm[wave][0];

  auto issue_tiles = [&](int k0, int buf) {
    // K tile: 32 consecutive rows of [SEQ,DIM] -> contiguous 16KB (1D)
    tdm_load_2d(lds_off(&ldsK[buf][0]),
                (unsigned long long)(const void*)(Kp + (size_t)k0 * DIM),
                32 * DIM, 1, 32 * DIM, 1, 32 * DIM);
    // V tile: 256 rows x 32 cols from Vt [DIM,SEQ], row stride SEQ
    tdm_load_2d(lds_off(&ldsV[buf][0]),
                (unsigned long long)(const void*)(Vp + (size_t)k0),
                SEQ, DIM, 32, DIM, SEQ);
  };

  if (wave == 0) issue_tiles(0, 0);

  for (int k0 = 0, it = 0; k0 < SEQ; k0 += 32, ++it) {
    int buf = it & 1;
    if (wave == 0) __builtin_amdgcn_s_wait_tensorcnt((short)0);
    __syncthreads();                       // tiles[buf] ready for all waves
    if (wave == 0 && k0 + 32 < SEQ) issue_tiles(k0 + 32, buf ^ 1);

    const unsigned short* kb = &ldsK[buf][0];
    const unsigned short* vb = &ldsV[buf][0];

    v8f s0, s1;
#pragma unroll
    for (int i = 0; i < 8; ++i) { s0[i] = 0.f; s1[i] = 0.f; }
#pragma unroll
    for (int c = 0; c < 8; ++c) {
      BF16Frag bk;
      const unsigned short* kr0 = kb + nl * DIM + c * 32 + hi * 8;
      bk.q[0] = *(const uint4*)(kr0);
      bk.q[1] = *(const uint4*)(kr0 + 16);
      s0 = __builtin_amdgcn_wmma_f32_16x16x32_bf16(
          false, aq[c].v, false, bk.v, (short)0, s0, false, false);
      const unsigned short* kr1 = kb + (16 + nl) * DIM + c * 32 + hi * 8;
      bk.q[0] = *(const uint4*)(kr1);
      bk.q[1] = *(const uint4*)(kr1 + 16);
      s1 = __builtin_amdgcn_wmma_f32_16x16x32_bf16(
          false, aq[c].v, false, bk.v, (short)0, s1, false, false);
    }

    // Srel gather + scale + online softmax
    float alpha[8];
#pragma unroll
    for (int i = 0; i < 8; ++i) {
      int r = q0 + hi * 8 + i;
      int cg0 = k0 + nl;
      float srel0 = (cg0 <= r)
          ? bf2f(QEp[(size_t)r * SEQ + (SEQ + cg0 - r - 1)])
          : (cg0 == r + 1) ? 0.f
          : bf2f(QEp[(size_t)(r + 1) * SEQ + (cg0 - r - 2)]);
      int cg1 = cg0 + 16;
      float srel1 = (cg1 <= r)
          ? bf2f(QEp[(size_t)r * SEQ + (SEQ + cg1 - r - 1)])
          : (cg1 == r + 1) ? 0.f
          : bf2f(QEp[(size_t)(r + 1) * SEQ + (cg1 - r - 2)]);
      float v0 = (s0[i] + srel0) * 0.0625f;   // 1/sqrt(256)
      float v1 = (s1[i] + srel1) * 0.0625f;
      float mt = fmaxf(v0, v1);
      mt = fmaxf(mt, __shfl_xor(mt, 1));
      mt = fmaxf(mt, __shfl_xor(mt, 2));
      mt = fmaxf(mt, __shfl_xor(mt, 4));
      mt = fmaxf(mt, __shfl_xor(mt, 8));
      float mnew = fmaxf(mrow[i], mt);
      alpha[i] = __expf(mrow[i] - mnew);
      mrow[i] = mnew;
      float p0 = __expf(v0 - mnew);
      float p1 = __expf(v1 - mnew);
      float ps = p0 + p1;
      ps += __shfl_xor(ps, 1);
      ps += __shfl_xor(ps, 2);
      ps += __shfl_xor(ps, 4);
      ps += __shfl_xor(ps, 8);
      srow[i] = srow[i] * alpha[i] + ps;
      s0[i] = p0; s1[i] = p1;
    }
#pragma unroll
    for (int t = 0; t < 16; ++t)
#pragma unroll
      for (int i = 0; i < 8; ++i) o[t][i] *= alpha[i];

    // C-layout -> A-layout for P via per-wave LDS stage
#pragma unroll
    for (int i = 0; i < 8; ++i) {
      pw[(hi * 8 + i) * 32 + nl]      = f2bf(s0[i]);
      pw[(hi * 8 + i) * 32 + 16 + nl] = f2bf(s1[i]);
    }
    __builtin_amdgcn_wave_barrier();
    asm volatile("s_wait_dscnt 0x0" ::: "memory");
    BF16Frag ap;
    ap.q[0] = *(const uint4*)(pw + nl * 32 + hi * 8);
    ap.q[1] = *(const uint4*)(pw + nl * 32 + 16 + hi * 8);
    __builtin_amdgcn_wave_barrier();

    // O += P @ V from LDS V tile (rows of 32 bf16)
#pragma unroll
    for (int t = 0; t < 16; ++t) {
      BF16Frag bv;
      const unsigned short* vr = vb + (t * 16 + nl) * 32 + hi * 8;
      bv.q[0] = *(const uint4*)(vr);
      bv.q[1] = *(const uint4*)(vr + 16);
      o[t] = __builtin_amdgcn_wmma_f32_16x16x32_bf16(
          false, ap.v, false, bv.v, (short)0, o[t], false, false);
    }
    __syncthreads();                       // all waves done with tiles[buf]
  }

#pragma unroll
  for (int i = 0; i < 8; ++i) {
    float inv = 1.f / srow[i];
    size_t rowoff = ((size_t)b * SEQ + q0 + hi * 8 + i) * DIM;
#pragma unroll
    for (int t = 0; t < 16; ++t)
      out[rowoff + t * 16 + nl] = o[t][i] * inv;
  }
}

// ---------------- host orchestration ----------------

extern "C" void kernel_launch(void* const* d_in, const int* in_sizes, int n_in,
                              void* d_out, int out_size, void* d_ws, size_t ws_size,
                              hipStream_t stream) {
  (void)in_sizes; (void)n_in; (void)out_size; (void)ws_size;
  const float* inQ = (const float*)d_in[0];
  const float* inK = (const float*)d_in[1];
  const float* inV = (const float*)d_in[2];
  const float* Wq  = (const float*)d_in[3];
  const float* Wk  = (const float*)d_in[4];
  const float* Wv  = (const float*)d_in[5];
  float* out = (float*)d_out;

  const size_t NELEM = (size_t)BATCH * SEQ * DIM;  // 4,194,304

  unsigned short* base  = (unsigned short*)d_ws;
  unsigned short* inQbf = base;                    // NELEM (reused as E later)
  unsigned short* inKbf = inQbf + NELEM;
  unsigned short* inVbf = inKbf + NELEM;
  unsigned short* wqt   = inVbf + NELEM;
  unsigned short* wkt   = wqt + DIM * DIM;
  unsigned short* wvt   = wkt + DIM * DIM;
  unsigned short* qbf   = wvt + DIM * DIM;
  unsigned short* kbf   = qbf + NELEM;
  unsigned short* vt    = kbf + NELEM;
  unsigned short* qe    = vt + NELEM;              // BATCH*SEQ*SEQ bf16
  unsigned short* ebf   = inQbf;                   // alias (inQbf dead by then)

  const int thr = 256;
  const int blkElem = (int)((NELEM + thr - 1) / thr);
  cvt_bf16_kernel<<<blkElem, thr, 0, stream>>>(inQ, inQbf, (int)NELEM);
  cvt_bf16_kernel<<<blkElem, thr, 0, stream>>>(inK, inKbf, (int)NELEM);
  cvt_bf16_kernel<<<blkElem, thr, 0, stream>>>(inV, inVbf, (int)NELEM);
  const int blkW = (DIM * DIM + thr - 1) / thr;
  cvt_wt_kernel<<<blkW, thr, 0, stream>>>(Wq, wqt);
  cvt_wt_kernel<<<blkW, thr, 0, stream>>>(Wk, wkt);
  cvt_wt_kernel<<<blkW, thr, 0, stream>>>(Wv, wvt);

  {  // projections: M=16384, N=256, K=256
    int waves = (16384 / 16) * (256 / 64);         // 4096
    dim3 grid(waves / 8, 1, 1);
    nt_gemm_bf16_kernel<<<grid, 256, 0, stream>>>(inQbf, wqt, qbf,
        16384, 256, 256, 0, 0, 0, 0);
    nt_gemm_bf16_kernel<<<grid, 256, 0, stream>>>(inKbf, wkt, kbf,
        16384, 256, 256, 0, 0, 0, 0);
    nt_gemm_bf16_kernel<<<grid, 256, 0, stream>>>(inVbf, wvt, vt,
        16384, 256, 256, 0, 0, 0, 1);              // store V transposed
  }

  make_e_kernel<<<blkElem, thr, 0, stream>>>(kbf, qbf, ebf, (int)NELEM);

  {  // QE[b] = Q[b] @ E[b]^T : M=N=2048, K=256, batched over z
    int waves = (SEQ / 16) * (SEQ / 64);           // 4096
    dim3 grid(waves / 8, 1, BATCH);
    nt_gemm_bf16_kernel<<<grid, 256, 0, stream>>>(qbf, ebf, qe,
        SEQ, SEQ, DIM,
        (long long)SEQ * DIM, (long long)SEQ * DIM, (long long)SEQ * SEQ, 0);
  }

  dim3 ag(SEQ / 128, BATCH, 1);
  rga_attn_kernel<<<ag, 256, 0, stream>>>(qbf, kbf, vt, qe, out);
}
